// CausalSelfAttention_61297773248680
// MI455X (gfx1250) — compile-verified, matrix operand order checked by reading, not run on hardware
//
#include <hip/hip_runtime.h>
#include <hip/hip_bf16.h>

// ---------------------------------------------------------------------------
// CausalSelfAttention (B=2, T=2048, C=1024, H=16, Dh=64) for gfx1250 / MI455X.
// All four matmuls run on v_wmma_f32_16x16x32_f16 (f16 operands, f32 accum).
// GEMM staging uses the Tensor Data Mover (tensor_load_to_lds + TENSORcnt),
// 6-arg builtin form (amdgpu-toolchain clang-23), else vector-load fallback.
// ---------------------------------------------------------------------------

typedef __attribute__((ext_vector_type(16))) _Float16 v16h;
typedef __attribute__((ext_vector_type(8)))  float    v8f;

union F16x8 { uint4 u; _Float16 h[8]; };

#if defined(__has_builtin)
#if __has_builtin(__builtin_amdgcn_tensor_load_to_lds) && \
    __has_builtin(__builtin_amdgcn_s_wait_tensorcnt)
#define USE_TDM 1
#endif
#endif
#ifndef USE_TDM
#define USE_TDM 0
#endif

#if USE_TDM
typedef unsigned int u32x4 __attribute__((ext_vector_type(4)));
typedef int          i32x8 __attribute__((ext_vector_type(8)));
typedef int          i32x4 __attribute__((ext_vector_type(4)));

// Issue one TDM 2D tile load: tile (tileRows x tileK f16) from a row-major
// (tensorRows x tensorK) f16 tensor into LDS at ldsOff.  HW padding inserts
// 4 DWORDs (16B) after every 32 DWORDs (64 f16 = one tile row), producing the
// LDS stride of 72 f16 that the WMMA fragment loads expect.
__device__ __forceinline__ void tdm_load_2d(const _Float16* gsrc,
                                            unsigned ldsOff, int tensorK,
                                            int tensorRows, int tileK,
                                            int tileRows) {
    unsigned long long ga = (unsigned long long)(uintptr_t)gsrc;
    u32x4 g0;
    g0[0] = 1u;                                       // count=1 (valid), load
    g0[1] = ldsOff;                                   // lds_addr (bytes)
    g0[2] = (unsigned)(ga & 0xFFFFFFFFu);             // global_addr[31:0]
    g0[3] = (unsigned)((ga >> 32) & 0x01FFFFFFu) | (2u << 30);  // type=2
    i32x8 g1;
    // data_size=1 (2B) <<16 | pad_enable<<20 | pad_interval=4 (32 DW) <<22
    //  | pad_amount=3 (4 DW) <<25
    g1[0] = (1 << 16) | (1 << 20) | (4 << 22) | (3 << 25);
    g1[1] = (tensorK & 0xFFFF) << 16;                 // tensor_dim0[15:0]
    g1[2] = ((tensorK >> 16) & 0xFFFF) | ((tensorRows & 0xFFFF) << 16);
    g1[3] = ((tensorRows >> 16) & 0xFFFF) | ((tileK & 0xFFFF) << 16);
    g1[4] = (tileRows & 0xFFFF);                      // tile_dim1 (tile_dim2=0)
    g1[5] = tensorK;                                  // tensor_dim0_stride lo32
    g1[6] = 0;
    g1[7] = 0;
    i32x4 z4 = {0, 0, 0, 0};
    i32x8 z8 = {0, 0, 0, 0, 0, 0, 0, 0};
    __builtin_amdgcn_tensor_load_to_lds(g0, g1, z4, z4, z8, 0);
}
#endif

// Load one 16x32 f16 A/B fragment chunk from (16B-aligned) LDS data.
// base must already include:  row(lane&15)*LD + kchunk*32 + (lane>>4)*8
// elems 0..7  = K(hi*8 + 0..7),  elems 8..15 = K(16 + hi*8 + 0..7)  (ISA 7.12.2)
__device__ __forceinline__ v16h load_frag(const _Float16* base) {
    union { v16h v; uint4 u[2]; } t;
    t.u[0] = *(const uint4*)(base);
    t.u[1] = *(const uint4*)(base + 16);
    return t.v;
}

// ---------------------------------------------------------------------------
// Elementwise converts
// ---------------------------------------------------------------------------
__global__ void f32_to_f16_kernel(const float* __restrict__ in,
                                  _Float16* __restrict__ out, int n) {
    int i = blockIdx.x * blockDim.x + threadIdx.x;
    if (i < n) out[i] = (_Float16)in[i];
}

// in: R x Ccols row-major f32  ->  out: Ccols x R row-major f16 (i.e. W^T)
__global__ void transpose_f32_to_f16_kernel(const float* __restrict__ in,
                                            _Float16* __restrict__ out,
                                            int R, int Ccols) {
    int i = blockIdx.x * blockDim.x + threadIdx.x;
    if (i >= R * Ccols) return;
    int r = i / Ccols, c = i % Ccols;
    out[(size_t)c * R + r] = (_Float16)in[(size_t)r * Ccols + c];
}

// ---------------------------------------------------------------------------
// WMMA GEMM:  C[M,N] = A[M,K] @ B[K,N],  B given transposed (Bt[N,K] row-major)
// Block: 128x128, BK=64, 8 waves (2x4), each wave 64x32 -> 16 WMMAs / K-step.
// ---------------------------------------------------------------------------
template <typename OutT>
__global__ __launch_bounds__(256) void gemm_wmma_kernel(
    const _Float16* __restrict__ A, const _Float16* __restrict__ Bt,
    OutT* __restrict__ C, int M, int N, int K) {
    constexpr int BM = 128, BN = 128, BK = 64;
    constexpr int LDA = BK + 8;                     // 72 f16 (matches TDM pad)
    __shared__ __align__(16) _Float16 sA[BM * LDA];
    __shared__ __align__(16) _Float16 sB[BN * LDA];

    const int tid  = threadIdx.x;
    const int lane = tid & 31, w = tid >> 5;
    const int wm = w >> 2, wn = w & 3;              // 2x4 wave grid
    const int lo = lane & 15, hi = lane >> 4;
    const int m0 = blockIdx.y * BM, n0 = blockIdx.x * BN;

    v8f acc[4][2] = {};

    for (int k0 = 0; k0 < K; k0 += BK) {
#if USE_TDM
        if (w == 0) {
            tdm_load_2d(&A[(size_t)m0 * K + k0], (unsigned)(uintptr_t)sA,
                        K, M, BK, BM);
            tdm_load_2d(&Bt[(size_t)n0 * K + k0], (unsigned)(uintptr_t)sB,
                        K, N, BK, BN);
            __builtin_amdgcn_s_wait_tensorcnt(0);
        }
#else
        // 128 rows x 8 16B-chunks per tile = 1024 chunks, 4 per thread per tile
#pragma unroll
        for (int i = 0; i < 4; ++i) {
            int idx = tid + i * 256;
            int r = idx >> 3, ch = idx & 7;
            *(uint4*)&sA[r * LDA + ch * 8] =
                *(const uint4*)&A[(size_t)(m0 + r) * K + k0 + ch * 8];
            *(uint4*)&sB[r * LDA + ch * 8] =
                *(const uint4*)&Bt[(size_t)(n0 + r) * K + k0 + ch * 8];
        }
#endif
        __syncthreads();

#pragma unroll
        for (int c = 0; c < 2; ++c) {
            v16h af[4], bf[2];
#pragma unroll
            for (int i = 0; i < 4; ++i)
                af[i] = load_frag(&sA[(wm * 64 + i * 16 + lo) * LDA + c * 32 + hi * 8]);
#pragma unroll
            for (int j = 0; j < 2; ++j)
                bf[j] = load_frag(&sB[(wn * 32 + j * 16 + lo) * LDA + c * 32 + hi * 8]);
#pragma unroll
            for (int i = 0; i < 4; ++i)
#pragma unroll
                for (int j = 0; j < 2; ++j)
                    acc[i][j] = __builtin_amdgcn_wmma_f32_16x16x32_f16(
                        false, af[i], false, bf[j], (short)0, acc[i][j],
                        false, false);
        }
        __syncthreads();
    }

    // C/D layout: VGPR r -> row r (lanes 0-15) or row 8+r (lanes 16-31)
#pragma unroll
    for (int i = 0; i < 4; ++i)
#pragma unroll
        for (int j = 0; j < 2; ++j) {
            int mb = m0 + wm * 64 + i * 16 + hi * 8;
            int nb = n0 + wn * 32 + j * 16 + lo;
#pragma unroll
            for (int r = 0; r < 8; ++r)
                C[(size_t)(mb + r) * N + nb] = (OutT)acc[i][j][r];
        }
}

// ---------------------------------------------------------------------------
// RoPE in-place on q,k halves of qkv (f16).  One thread per (b,t,part,h,j) pair.
// ---------------------------------------------------------------------------
__global__ void rope_kernel(_Float16* __restrict__ qkv, int B, int T, int H,
                            int Cdim) {
    const int half = 32;                       // Dh/2
    int idx = blockIdx.x * blockDim.x + threadIdx.x;
    int total = B * T * 2 * H * half;
    if (idx >= total) return;
    int j = idx % half;  idx /= half;
    int h = idx % H;     idx /= H;
    int part = idx % 2;  idx /= 2;
    int t = idx % T;
    int b = idx / T;

    size_t row = (size_t)(b * T + t) * (3 * Cdim);
    size_t col = (size_t)part * Cdim + h * 64 + j;
    float x1 = (float)qkv[row + col];
    float x2 = (float)qkv[row + col + half];
    float inv = __powf(10000.0f, -(float)j * (1.0f / 32.0f));
    float sn, cs;
    __sincosf((float)t * inv, &sn, &cs);
    qkv[row + col]        = (_Float16)(x1 * cs - x2 * sn);
    qkv[row + col + half] = (_Float16)(x2 * cs + x1 * sn);
}

// ---------------------------------------------------------------------------
// Flash attention, causal. Block = (qb, h, b): 64 query rows, 4 waves x 16 rows.
// S = Q K^T (WMMA), online softmax in C-layout, O += P V (WMMA, P via LDS).
// ---------------------------------------------------------------------------
__global__ __launch_bounds__(128) void flash_attn_kernel(
    const _Float16* __restrict__ qkv, _Float16* __restrict__ Y,
    int B, int T, int H, int Cdim) {
    constexpr int Dh = 64, LD = 72;            // padded LDS stride (f16)
    __shared__ __align__(16) _Float16 sQ[64 * LD];
    __shared__ __align__(16) _Float16 sK[64 * LD];
    __shared__ __align__(16) _Float16 sVt[64 * LD];
    __shared__ __align__(16) _Float16 sP[4 * 16 * LD];

    const int qb = blockIdx.x, h = blockIdx.y, b = blockIdx.z;
    const int tid = threadIdx.x, lane = tid & 31, w = tid >> 5;
    const int lo = lane & 15, hi = lane >> 4;
    const int row3 = 3 * Cdim;
    const int qcol = h * Dh, kcol = Cdim + h * Dh, vcol = 2 * Cdim + h * Dh;
    const int qrow0 = qb * 64;

    // Stage Q tile (64x64 f16)
#pragma unroll
    for (int i = 0; i < 4; ++i) {
        int idx = tid + i * 128;
        int r = idx >> 3, ch = idx & 7;
        *(uint4*)&sQ[r * LD + ch * 8] =
            *(const uint4*)&qkv[(size_t)(b * T + qrow0 + r) * row3 + qcol + ch * 8];
    }
    __syncthreads();
    v16h qf[2];
#pragma unroll
    for (int c = 0; c < 2; ++c)
        qf[c] = load_frag(&sQ[(w * 16 + lo) * LD + c * 32 + hi * 8]);

    v8f o[4] = {};
    float mrow[8], lrow[8];
#pragma unroll
    for (int r = 0; r < 8; ++r) { mrow[r] = -1e30f; lrow[r] = 0.0f; }
    const float scale = 0.125f;                // 1/sqrt(64)

    for (int kb = 0; kb <= qb; ++kb) {
        __syncthreads();                       // protect sK/sVt reuse
        // Stage K tile (row-major) and V tile transposed
#pragma unroll
        for (int i = 0; i < 4; ++i) {
            int idx = tid + i * 128;
            int r = idx >> 3, ch = idx & 7;
            size_t g = (size_t)(b * T + kb * 64 + r) * row3;
            *(uint4*)&sK[r * LD + ch * 8] = *(const uint4*)&qkv[g + kcol + ch * 8];
            F16x8 vv; vv.u = *(const uint4*)&qkv[g + vcol + ch * 8];
#pragma unroll
            for (int j = 0; j < 8; ++j) sVt[(ch * 8 + j) * LD + r] = vv.h[j];
        }
        __syncthreads();

        // S = Q @ K^T  (B-operand = K rows, since B^T = K_blk row-major)
        v8f s[4];
#pragma unroll
        for (int nt = 0; nt < 4; ++nt) {
            v8f z = {};
            v16h kf0 = load_frag(&sK[(nt * 16 + lo) * LD + 0 * 32 + hi * 8]);
            v16h kf1 = load_frag(&sK[(nt * 16 + lo) * LD + 1 * 32 + hi * 8]);
            z = __builtin_amdgcn_wmma_f32_16x16x32_f16(false, qf[0], false, kf0,
                                                       (short)0, z, false, false);
            z = __builtin_amdgcn_wmma_f32_16x16x32_f16(false, qf[1], false, kf1,
                                                       (short)0, z, false, false);
            s[nt] = z;
        }

        // scale + causal mask (only diagonal block needs it)
        const int qrow_lane = qrow0 + w * 16 + hi * 8;
        const bool diag = (kb == qb);
#pragma unroll
        for (int nt = 0; nt < 4; ++nt) {
            int kg = kb * 64 + nt * 16 + lo;
#pragma unroll
            for (int r = 0; r < 8; ++r) {
                float v = s[nt][r] * scale;
                if (diag && kg > qrow_lane + r) v = -1e30f;
                s[nt][r] = v;
            }
        }

        // online softmax: row reductions across 16 lanes of each half-wave
        float mnew[8], corr[8];
#pragma unroll
        for (int r = 0; r < 8; ++r) {
            float mx = s[0][r];
#pragma unroll
            for (int nt = 1; nt < 4; ++nt) mx = fmaxf(mx, s[nt][r]);
            for (int off = 1; off < 16; off <<= 1)
                mx = fmaxf(mx, __shfl_xor(mx, off, 32));
            mnew[r] = fmaxf(mrow[r], mx);
            corr[r] = __expf(mrow[r] - mnew[r]);
        }
#pragma unroll
        for (int nt = 0; nt < 4; ++nt)
#pragma unroll
            for (int r = 0; r < 8; ++r) o[nt][r] *= corr[r];
#pragma unroll
        for (int r = 0; r < 8; ++r) {
            float sum = 0.0f;
#pragma unroll
            for (int nt = 0; nt < 4; ++nt) {
                float p = __expf(s[nt][r] - mnew[r]);
                s[nt][r] = p;
                sum += p;
            }
            for (int off = 1; off < 16; off <<= 1)
                sum += __shfl_xor(sum, off, 32);
            lrow[r] = lrow[r] * corr[r] + sum;
            mrow[r] = mnew[r];
        }

        // P: C-layout regs -> per-wave LDS (16x64 f16) -> A-fragment layout
        _Float16* pw = &sP[w * 16 * LD];
#pragma unroll
        for (int nt = 0; nt < 4; ++nt)
#pragma unroll
            for (int r = 0; r < 8; ++r)
                pw[(hi * 8 + r) * LD + nt * 16 + lo] = (_Float16)s[nt][r];
        v16h pf0 = load_frag(&pw[lo * LD + 0 * 32 + hi * 8]);
        v16h pf1 = load_frag(&pw[lo * LD + 1 * 32 + hi * 8]);

        // O += P @ V  (B^T = V^T row-major in sVt)
#pragma unroll
        for (int nt = 0; nt < 4; ++nt) {
            v16h vf0 = load_frag(&sVt[(nt * 16 + lo) * LD + 0 * 32 + hi * 8]);
            v16h vf1 = load_frag(&sVt[(nt * 16 + lo) * LD + 1 * 32 + hi * 8]);
            o[nt] = __builtin_amdgcn_wmma_f32_16x16x32_f16(false, pf0, false, vf0,
                                                           (short)0, o[nt], false, false);
            o[nt] = __builtin_amdgcn_wmma_f32_16x16x32_f16(false, pf1, false, vf1,
                                                           (short)0, o[nt], false, false);
        }
    }

    // normalize and store y (f16)
#pragma unroll
    for (int nt = 0; nt < 4; ++nt)
#pragma unroll
        for (int r = 0; r < 8; ++r) {
            float val = o[nt][r] / lrow[r];
            int grow = b * T + qrow0 + w * 16 + hi * 8 + r;
            Y[(size_t)grow * Cdim + h * Dh + nt * 16 + lo] = (_Float16)val;
        }
}

// ---------------------------------------------------------------------------
// Launch
// ---------------------------------------------------------------------------
extern "C" void kernel_launch(void* const* d_in, const int* in_sizes, int n_in,
                              void* d_out, int out_size, void* d_ws, size_t ws_size,
                              hipStream_t stream) {
    (void)in_sizes; (void)n_in; (void)out_size; (void)ws_size;
    const float* x     = (const float*)d_in[0];
    const float* Wqkv  = (const float*)d_in[1];
    const float* Wproj = (const float*)d_in[2];
    float* out = (float*)d_out;

    const int B = 2, T = 2048, C = 1024, H = 16;
    const int M = B * T;                        // 4096

    // workspace carve (f16): ~48 MB total
    _Float16* xh     = (_Float16*)d_ws;                     // M x C
    _Float16* wqkvt  = xh    + (size_t)M * C;               // 3C x C
    _Float16* wprojt = wqkvt + (size_t)3 * C * C;           // C x C
    _Float16* qkvh   = wprojt + (size_t)C * C;              // M x 3C
    _Float16* yh     = qkvh  + (size_t)M * 3 * C;           // M x C

    // 1) converts
    {
        int n = M * C;
        f32_to_f16_kernel<<<(n + 255) / 256, 256, 0, stream>>>(x, xh, n);
        int nq = C * 3 * C;
        transpose_f32_to_f16_kernel<<<(nq + 255) / 256, 256, 0, stream>>>(
            Wqkv, wqkvt, C, 3 * C);
        int np = C * C;
        transpose_f32_to_f16_kernel<<<(np + 255) / 256, 256, 0, stream>>>(
            Wproj, wprojt, C, C);
    }
    // 2) qkv = x @ W_qkv   (M=4096, N=3072, K=1024) -> f16
    gemm_wmma_kernel<_Float16><<<dim3(3 * C / 128, M / 128), 256, 0, stream>>>(
        xh, wqkvt, qkvh, M, 3 * C, C);
    // 3) RoPE on q,k
    {
        int total = B * T * 2 * H * 32;
        rope_kernel<<<(total + 255) / 256, 256, 0, stream>>>(qkvh, B, T, H, C);
    }
    // 4) flash attention -> yh (f16)
    flash_attn_kernel<<<dim3(T / 64, H, B), 128, 0, stream>>>(qkvh, yh, B, T, H, C);
    // 5) out = y @ W_proj  (M=4096, N=1024, K=1024) -> f32
    gemm_wmma_kernel<float><<<dim3(C / 128, M / 128), 256, 0, stream>>>(
        yh, wprojt, out, M, C, C);
}
